// GraphProcessor_23450521436179
// MI455X (gfx1250) — matrix operand order repaired and checked by reading, not compiled, not run on hardware
//
#include <hip/hip_runtime.h>

// ---------------- problem constants (match reference) ----------------
constexpr int NN = 10000;    // nodes
constexpr int NE = 160000;   // edges (before self loops)
constexpr int ET = NE + NN;  // edges incl self loops
constexpr int DI = 256, DH = 512, DO = 128;
constexpr int NG = 8;

// ---------------- WMMA types ----------------
typedef __attribute__((ext_vector_type(16))) __bf16 v16bf;
typedef __attribute__((ext_vector_type(8)))  float  v8f;

union BFrag { v16bf v; uint4 q[2]; };

__device__ inline unsigned short f32_to_bf16_bits(float f) {
  unsigned u = __float_as_uint(f);
  return (unsigned short)((u + 0x7FFFu + ((u >> 16) & 1u)) >> 16);  // RNE
}

__device__ inline void atomicMaxF(float* addr, float v) {
  // float max via monotone int mapping (sign-split)
  if (v >= 0.0f) atomicMax((int*)addr, __float_as_int(v));
  else           atomicMin((unsigned int*)addr, __float_as_uint(v));
}

// ---------------- f32 -> bf16 conversion ----------------
__global__ void k_cvt_bf16(const float* __restrict__ in, unsigned short* __restrict__ out, int n) {
  int i = blockIdx.x * blockDim.x + threadIdx.x;
  if (i < n) out[i] = f32_to_bf16_bits(in[i]);
}

// ---------------- WMMA GEMM: C[M,N] = A[M,K] @ B[K,N], bf16 in, f32 out -----
// One wave computes a 16(M) x 64(N) strip: 4 accumulators, 4 wmma per K-step.
__global__ void k_gemm_wmma(const unsigned short* __restrict__ Abits,
                            const unsigned short* __restrict__ Bbits,
                            float* __restrict__ C,
                            int K, int N, int nWaveTiles) {
  const __bf16* A = (const __bf16*)Abits;
  const __bf16* B = (const __bf16*)Bbits;
  int wid  = (blockIdx.x * blockDim.x + threadIdx.x) >> 5;
  if (wid >= nWaveTiles) return;
  int lane = threadIdx.x & 31;
  int nt4  = N >> 6;                 // 64-col strips
  int mt   = wid / nt4;
  int nt   = wid % nt4;
  int g    = lane >> 4;              // half-wave
  int row  = mt * 16 + (lane & 15);  // A row for this lane
  int bkr  = (lane & 15) + (g << 4); // B k-row within 32-k tile for this lane

  v8f c0 = 0.0f, c1 = 0.0f, c2 = 0.0f, c3 = 0.0f;

  for (int kb = 0; kb < K; kb += 32) {
    // A fragment: 16-bit A 16x32 layout -> two contiguous 8-elem chunks per lane
    BFrag a;
    const __bf16* ap = A + (size_t)row * K + kb + g * 8;
    a.q[0] = *(const uint4*)(ap);
    a.q[1] = *(const uint4*)(ap + 16);
    // B fragments: lane holds column n = lane%16, 16 k values in the same order as A
    const __bf16* bp = B + (size_t)kb * N + nt * 64 + (lane & 15);
    BFrag b0, b1, b2, b3;
    for (int s = 0; s < 16; ++s) {
      int kk = (s < 8) ? (g * 8 + s) : (16 + g * 8 + (s - 8));
      const __bf16* bq = bp + (size_t)kk * N;
      b0.v[s] = bq[0]; b1.v[s] = bq[16]; b2.v[s] = bq[32]; b3.v[s] = bq[48];
    }

    c0 = __builtin_amdgcn_wmma_f32_16x16x32_bf16(false, a.v, false, b0.v, (short)0, c0, false, false);
    c1 = __builtin_amdgcn_wmma_f32_16x16x32_bf16(false, a.v, false, b1.v, (short)0, c1, false, false);
    c2 = __builtin_amdgcn_wmma_f32_16x16x32_bf16(false, a.v, false, b2.v, (short)0, c2, false, false);
    c3 = __builtin_amdgcn_wmma_f32_16x16x32_bf16(false, a.v, false, b3.v, (short)0, c3, false, false);
  }

  // C/D layout: VGPR r -> M = r + 8*g, N = lane%16
  int colb = nt * 64 + (lane & 15);
  for (int r = 0; r < 8; ++r) {
    int orow = mt * 16 + r + 8 * g;
    float* cp = C + (size_t)orow * N + colb;
    cp[0]  = c0[r];
    cp[16] = c1[r];
    cp[32] = c2[r];
    cp[48] = c3[r];
  }
}

// ---------------- per-node attention score dots ----------------
__global__ void k_scores(const float* __restrict__ h, const float* __restrict__ as,
                         const float* __restrict__ ad, float* __restrict__ ss,
                         float* __restrict__ sd, int nc) {
  int w = (blockIdx.x * blockDim.x + threadIdx.x) >> 5;
  int lane = threadIdx.x & 31;
  if (w >= NN) return;
  const float* hp = h + (size_t)w * nc;
  float s0 = 0.f, s1 = 0.f;
  for (int c = lane; c < nc; c += 32) { float v = hp[c]; s0 += v * as[c]; s1 += v * ad[c]; }
  for (int o = 16; o; o >>= 1) { s0 += __shfl_xor(s0, o, 32); s1 += __shfl_xor(s1, o, 32); }
  if (lane == 0) { ss[w] = s0; sd[w] = s1; }
}

// ---------------- node arrays init (max=-inf, denom=0) ----------------
__global__ void k_init_nodes(float* __restrict__ mx, float* __restrict__ dn) {
  int i = blockIdx.x * blockDim.x + threadIdx.x;
  if (i < NN) { mx[i] = __int_as_float(0xff800000); dn[i] = 0.f; }
}

// ---------------- edge pass 1: leaky_relu + segment max ----------------
__global__ void k_edge_max(const long long* __restrict__ ei, const float* __restrict__ ss,
                           const float* __restrict__ sd, float* __restrict__ ev,
                           float* __restrict__ mx) {
  int t = blockIdx.x * blockDim.x + threadIdx.x;
  if (t >= ET) return;
  int s, d;
  if (t < NE) { s = (int)ei[t]; d = (int)ei[NE + t]; } else { s = d = t - NE; }
  float v = ss[s] + sd[d];
  v = v > 0.f ? v : 0.2f * v;           // leaky relu, slope 0.2
  ev[t] = v;
  atomicMaxF(mx + d, v);
}

// ---------------- edge pass 2: exp + segment sum ----------------
__global__ void k_edge_exp(const long long* __restrict__ ei, float* __restrict__ ev,
                           const float* __restrict__ mx, float* __restrict__ dn) {
  int t = blockIdx.x * blockDim.x + threadIdx.x;
  if (t >= ET) return;
  int d = (t < NE) ? (int)ei[NE + t] : (t - NE);
  float v = __expf(ev[t] - mx[d]);
  ev[t] = v;
  atomicAdd(dn + d, v);
}

// ---------------- output init with bias ----------------
__global__ void k_init_bias(float* __restrict__ out, const float* __restrict__ b, int nc, int total) {
  int i = blockIdx.x * blockDim.x + threadIdx.x;
  if (i < total) out[i] = b[i % nc];
}

// ---------------- edge pass 3: scatter alpha * h[src] into out[dst] -------
__global__ void k_edge_scatter(const long long* __restrict__ ei, const float* __restrict__ ev,
                               const float* __restrict__ dn, const float* __restrict__ h,
                               float* __restrict__ out, int nc) {
  int w = (blockIdx.x * blockDim.x + threadIdx.x) >> 5;
  int lane = threadIdx.x & 31;
  if (w >= ET) return;
  int s, d;
  if (w < NE) { s = (int)ei[w]; d = (int)ei[NE + w]; } else { s = d = w - NE; }
  float coef = ev[w] / fmaxf(dn[d], 1e-16f);
  const float* hp = h + (size_t)s * nc;
  float* op = out + (size_t)d * nc;
  for (int c = lane; c < nc; c += 32) atomicAdd(op + c, coef * hp[c]);
}

// ---------------- relu in-place + bf16 copy for next layer ----------------
__global__ void k_relu_cvt(float* __restrict__ io, unsigned short* __restrict__ ob, int n) {
  int i = blockIdx.x * blockDim.x + threadIdx.x;
  if (i < n) {
    float v = fmaxf(io[i], 0.f);
    io[i] = v;
    ob[i] = f32_to_bf16_bits(v);
  }
}

// ---------------- mean pool ----------------
__global__ void k_pool_zero(float* __restrict__ ps, float* __restrict__ pc) {
  int i = blockIdx.x * blockDim.x + threadIdx.x;
  if (i < NG * DO) ps[i] = 0.f;
  if (i < NG)      pc[i] = 0.f;
}

__global__ void k_pool_accum(const float* __restrict__ h, const long long* __restrict__ batch,
                             float* __restrict__ ps, float* __restrict__ pc) {
  int w = (blockIdx.x * blockDim.x + threadIdx.x) >> 5;
  int lane = threadIdx.x & 31;
  if (w >= NN) return;
  int g = (int)batch[w];
  const float* hp = h + (size_t)w * DO;
  float* op = ps + (size_t)g * DO;
  for (int c = lane; c < DO; c += 32) atomicAdd(op + c, hp[c]);
  if (lane == 0) atomicAdd(pc + g, 1.f);
}

__global__ void k_pool_final(const float* __restrict__ ps, const float* __restrict__ pc,
                             float* __restrict__ out) {
  int i = blockIdx.x * blockDim.x + threadIdx.x;
  if (i < NG * DO) out[i] = ps[i] / fmaxf(pc[i >> 7], 1.f);  // DO==128
}

// =====================================================================
extern "C" void kernel_launch(void* const* d_in, const int* in_sizes, int n_in,
                              void* d_out, int out_size, void* d_ws, size_t ws_size,
                              hipStream_t stream) {
  const float*     x   = (const float*)d_in[0];
  const long long* ei  = (const long long*)d_in[1];  // int64 [2, NE]
  const long long* bat = (const long long*)d_in[3];  // int64 [NN]
  const float* W1 = (const float*)d_in[4];
  const float* as1 = (const float*)d_in[5], *ad1 = (const float*)d_in[6], *b1 = (const float*)d_in[7];
  const float* W2 = (const float*)d_in[8];
  const float* as2 = (const float*)d_in[9], *ad2 = (const float*)d_in[10], *b2 = (const float*)d_in[11];
  const float* W3 = (const float*)d_in[12];
  const float* as3 = (const float*)d_in[13], *ad3 = (const float*)d_in[14], *b3 = (const float*)d_in[15];
  float* out = (float*)d_out;

  // ---- workspace bump allocator (256B aligned) ----
  size_t off = 0;
  auto alloc = [&](size_t bytes) -> char* {
    char* p = (char*)d_ws + off;
    off += (bytes + 255) & ~size_t(255);
    return p;
  };
  unsigned short* xb  = (unsigned short*)alloc((size_t)NN * DI * 2);
  unsigned short* w1b = (unsigned short*)alloc((size_t)DI * DH * 2);
  unsigned short* w2b = (unsigned short*)alloc((size_t)DH * DH * 2);
  unsigned short* w3b = (unsigned short*)alloc((size_t)DH * DO * 2);
  float* G   = (float*)alloc((size_t)NN * DH * 4);   // shared GEMM output (g1/g2/g3)
  float* o1  = (float*)alloc((size_t)NN * DH * 4);   // layer1 out; reused as o3
  unsigned short* o1b = (unsigned short*)alloc((size_t)NN * DH * 2); // reused as o2b
  float* o2  = (float*)alloc((size_t)NN * DH * 4);
  float* ss  = (float*)alloc((size_t)NN * 4);
  float* sd  = (float*)alloc((size_t)NN * 4);
  float* mx  = (float*)alloc((size_t)NN * 4);
  float* dn  = (float*)alloc((size_t)NN * 4);
  float* ev  = (float*)alloc((size_t)ET * 4);
  float* ps  = (float*)alloc((size_t)NG * DO * 4);
  float* pc  = (float*)alloc((size_t)NG * 4);
  float* o3  = o1;                                   // [NN, DO] fits in o1
  unsigned short* o2b = o1b;                         // safe: o1b consumed by layer2 GEMM

  const int T = 256;
  auto cdiv = [](long long a, long long b) { return (int)((a + b - 1) / b); };

  // ---- convert inputs/weights to bf16 ----
  k_cvt_bf16<<<cdiv((long long)NN * DI, T), T, 0, stream>>>(x,  xb,  NN * DI);
  k_cvt_bf16<<<cdiv((long long)DI * DH, T), T, 0, stream>>>(W1, w1b, DI * DH);
  k_cvt_bf16<<<cdiv((long long)DH * DH, T), T, 0, stream>>>(W2, w2b, DH * DH);
  k_cvt_bf16<<<cdiv((long long)DH * DO, T), T, 0, stream>>>(W3, w3b, DH * DO);

  const int eb = cdiv(ET, T);             // edge-per-thread blocks
  const int sb = cdiv((long long)ET * 32, T);  // edge-per-wave blocks
  const int nb = cdiv((long long)NN * 32, T);  // node-per-wave blocks

  // ================= layer 1: 256 -> 512 =================
  {
    int waves = (NN / 16) * (DH / 64);
    k_gemm_wmma<<<cdiv((long long)waves * 32, T), T, 0, stream>>>(xb, w1b, G, DI, DH, waves);
    k_scores<<<nb, T, 0, stream>>>(G, as1, ad1, ss, sd, DH);
    k_init_nodes<<<cdiv(NN, T), T, 0, stream>>>(mx, dn);
    k_edge_max<<<eb, T, 0, stream>>>(ei, ss, sd, ev, mx);
    k_edge_exp<<<eb, T, 0, stream>>>(ei, ev, mx, dn);
    k_init_bias<<<cdiv((long long)NN * DH, T), T, 0, stream>>>(o1, b1, DH, NN * DH);
    k_edge_scatter<<<sb, T, 0, stream>>>(ei, ev, dn, G, o1, DH);
    k_relu_cvt<<<cdiv((long long)NN * DH, T), T, 0, stream>>>(o1, o1b, NN * DH);
  }
  // ================= layer 2: 512 -> 512 =================
  {
    int waves = (NN / 16) * (DH / 64);
    k_gemm_wmma<<<cdiv((long long)waves * 32, T), T, 0, stream>>>(o1b, w2b, G, DH, DH, waves);
    k_scores<<<nb, T, 0, stream>>>(G, as2, ad2, ss, sd, DH);
    k_init_nodes<<<cdiv(NN, T), T, 0, stream>>>(mx, dn);
    k_edge_max<<<eb, T, 0, stream>>>(ei, ss, sd, ev, mx);
    k_edge_exp<<<eb, T, 0, stream>>>(ei, ev, mx, dn);
    k_init_bias<<<cdiv((long long)NN * DH, T), T, 0, stream>>>(o2, b2, DH, NN * DH);
    k_edge_scatter<<<sb, T, 0, stream>>>(ei, ev, dn, G, o2, DH);
    k_relu_cvt<<<cdiv((long long)NN * DH, T), T, 0, stream>>>(o2, o2b, NN * DH);
  }
  // ================= layer 3: 512 -> 128 (no relu) =================
  {
    int waves = (NN / 16) * (DO / 64);
    k_gemm_wmma<<<cdiv((long long)waves * 32, T), T, 0, stream>>>(o2b, w3b, G, DH, DO, waves);
    k_scores<<<nb, T, 0, stream>>>(G, as3, ad3, ss, sd, DO);
    k_init_nodes<<<cdiv(NN, T), T, 0, stream>>>(mx, dn);
    k_edge_max<<<eb, T, 0, stream>>>(ei, ss, sd, ev, mx);
    k_edge_exp<<<eb, T, 0, stream>>>(ei, ev, mx, dn);
    k_init_bias<<<cdiv((long long)NN * DO, T), T, 0, stream>>>(o3, b3, DO, NN * DO);
    k_edge_scatter<<<sb, T, 0, stream>>>(ei, ev, dn, G, o3, DO);
  }
  // ================= mean pool =================
  k_pool_zero<<<cdiv(NG * DO, T), T, 0, stream>>>(ps, pc);
  k_pool_accum<<<nb, T, 0, stream>>>(o3, bat, ps, pc);
  k_pool_final<<<cdiv(NG * DO, T), T, 0, stream>>>(ps, pc, out);

  (void)in_sizes; (void)n_in; (void)out_size; (void)ws_size;
}
